// OFTInjectedLinear_3221225472410
// MI455X (gfx1250) — matrix operand order repaired and checked by reading, not compiled, not run on hardware
//
#include <hip/hip_runtime.h>
#include <hip/hip_bf16.h>

typedef __attribute__((ext_vector_type(2))) float v2f;
typedef __attribute__((ext_vector_type(8))) float v8f;
typedef __attribute__((ext_vector_type(16))) __bf16 v16bf;

#define SEQ    4096
#define IN_F   4096
#define OUT_F  4096
#define POOL   5
#define RB     4
#define DD     1024

// d_ws layout (float units)
#define WS_B     ((size_t)0)                // 2*4*1024*1024 = 8388608 floats
#define WS_MEAN  ((size_t)8388608)          // 4096
#define WS_OMG   (WS_MEAN + 4096)           // 8
#define WS_INT   (WS_OMG + 8)               // 4 ints: sel0, sel1, k, pad
#define WS_SCL   (WS_INT + 4)               // 8 floats: 0.5*scale per (t,b)
#define WS_FAC   (WS_SCL + 8)               // 2*4096 per-task diag factors
#define WS_DVEC  (WS_FAC + 8192)            // 4096 final diagonal

__device__ __forceinline__ v8f wmma_f32(v2f a, v2f b, v8f c) {
  return __builtin_amdgcn_wmma_f32_16x16x4_f32(false, a, false, b, (short)0, c,
                                               false, false);
}
__device__ __forceinline__ v8f wmma_bf16(v16bf a, v16bf b, v8f c) {
  return __builtin_amdgcn_wmma_f32_16x16x32_bf16(false, a, false, b, (short)0,
                                                 c, false, false);
}

// f32 WMMA 128x128 x K=16 inner step (used by B = A@A).  Xs/Ws are [128][20].
__device__ __forceinline__ void mma_128x128_k16(const float* Xs, const float* Ws,
                                                v8f acc[2][4], int wm, int wn,
                                                int lane) {
  const int mrow = lane & 15;
  const int khalf = (lane >> 4) << 1;
#pragma unroll
  for (int kk = 0; kk < 16; kk += 4) {
    v2f a[2], b[4];
#pragma unroll
    for (int im = 0; im < 2; ++im)
      a[im] = *(const v2f*)&Xs[(wm + im * 16 + mrow) * 20 + kk + khalf];
#pragma unroll
    for (int in = 0; in < 4; ++in)
      b[in] = *(const v2f*)&Ws[(wn + in * 16 + mrow) * 20 + kk + khalf];
#pragma unroll
    for (int im = 0; im < 2; ++im)
#pragma unroll
      for (int in = 0; in < 4; ++in)
        acc[im][in] = wmma_f32(a[im], b[in], acc[im][in]);
  }
}

// ---- K1: column mean over sequence -----------------------------------------
__global__ __launch_bounds__(256) void mean_kernel(const float* __restrict__ X,
                                                   float* __restrict__ ws) {
  const int i = blockIdx.x * 256 + threadIdx.x;
  float acc = 0.f;
  for (int s = 0; s < SEQ; ++s) acc += X[(size_t)s * IN_F + i];
  ws[WS_MEAN + i] = acc * (1.0f / (float)SEQ);
}

// ---- K2: route logits, softmax, top-k --------------------------------------
__global__ __launch_bounds__(256) void route_topk_kernel(
    const float* __restrict__ route, const int* __restrict__ task_id_p,
    float* __restrict__ ws) {
  __shared__ float red[POOL][256];
  const int tid = threadIdx.x;
  const int task_id = *task_id_p;
  const float* mean = ws + WS_MEAN;
  const float* rt = route + (size_t)(task_id - 1) * IN_F * POOL;
  float part[POOL] = {0.f, 0.f, 0.f, 0.f, 0.f};
  for (int i = tid; i < IN_F; i += 256) {
    const float m = mean[i];
#pragma unroll
    for (int p = 0; p < POOL; ++p) part[p] += m * rt[(size_t)i * POOL + p];
  }
#pragma unroll
  for (int p = 0; p < POOL; ++p) red[p][tid] = part[p];
  __syncthreads();
  for (int s = 128; s > 0; s >>= 1) {
    if (tid < s)
#pragma unroll
      for (int p = 0; p < POOL; ++p) red[p][tid] += red[p][tid + s];
    __syncthreads();
  }
  if (tid == 0) {
    float lg[POOL];
#pragma unroll
    for (int p = 0; p < POOL; ++p) lg[p] = red[p][0];
    float mx = lg[0];
#pragma unroll
    for (int p = 1; p < POOL; ++p) mx = fmaxf(mx, lg[p]);
    float ex[POOL], sum = 0.f;
#pragma unroll
    for (int p = 0; p < POOL; ++p) { ex[p] = __expf(lg[p] - mx); sum += ex[p]; }
    float* omg = ws + WS_OMG;
#pragma unroll
    for (int p = 0; p < POOL; ++p) omg[p] = ex[p] / sum;
    const int limit = task_id < POOL ? task_id : POOL;
    const int k = task_id < 2 ? task_id : 2;
    int a0 = 0;
    for (int p = 1; p < limit; ++p)
      if (omg[p] > omg[a0]) a0 = p;          // ties -> lower index, like top_k
    int a1 = -1;
    for (int p = 0; p < limit; ++p) {
      if (p == a0) continue;
      if (a1 < 0 || omg[p] > omg[a1]) a1 = p;
    }
    if (a1 < 0) a1 = a0;
    int* ip = (int*)(ws + WS_INT);
    ip[0] = a0; ip[1] = a1; ip[2] = k;
  }
}

// ---- K3: per-(task,block) Frobenius norm -> COFT scale ---------------------
__global__ __launch_bounds__(256) void norm_scale_kernel(
    const float* __restrict__ lora_down, float* __restrict__ ws) {
  __shared__ float red[256];
  const int tid = threadIdx.x;
  const int p = blockIdx.x;
  const int t = p >> 2, b = p & 3;
  const int* ip = (const int*)(ws + WS_INT);
  const int k = ip[2];
  if (t >= k) { if (tid == 0) ws[WS_SCL + p] = 0.f; return; }
  const int j = ip[t];
  const float* R = lora_down + (size_t)j * RB * DD * DD + (size_t)b * DD * DD;
  float acc = 0.f;
  for (int idx = tid; idx < DD * DD; idx += 256) {
    const float r = R[idx];
    acc += r * r;
  }
  red[tid] = acc;
  __syncthreads();
  for (int s = 128; s > 0; s >>= 1) {
    if (tid < s) red[tid] += red[tid + s];
    __syncthreads();
  }
  if (tid == 0) {
    const float nrm = sqrtf(red[0]);
    const float eps = 1e-5f * (float)(DD * DD) / sqrtf((float)RB);  // 5.24288
    const float scale = (nrm <= eps) ? 1.0f : eps / fmaxf(nrm, 1e-12f);
    ws[WS_SCL + p] = 0.5f * scale;
  }
}

// ---- K4: B = A @ A per (task,block), f32 WMMA, A built on the fly ----------
__global__ __launch_bounds__(256) void gemm_B_kernel(
    const float* __restrict__ lora_down, float* __restrict__ ws) {
  __shared__ float Xs[128 * 20];
  __shared__ float Ws[128 * 20];
  const int tid = threadIdx.x;
  const int wave = tid >> 5, lane = tid & 31;
  const int wm = (wave & 3) * 32;
  const int wn = (wave >> 2) * 64;
  const int p = blockIdx.z, t = p >> 2, b = p & 3;
  const int* ip = (const int*)(ws + WS_INT);
  if (t >= ip[2]) return;
  const int j = ip[t];
  const float s = ws[WS_SCL + p];
  const float* R = lora_down + (size_t)j * RB * DD * DD + (size_t)b * DD * DD;
  float* Bout = ws + WS_B + (size_t)p * DD * DD;
  const int i0 = blockIdx.y * 128;
  const int j0 = blockIdx.x * 128;

  v8f acc[2][4];
#pragma unroll
  for (int im = 0; im < 2; ++im)
#pragma unroll
    for (int in = 0; in < 4; ++in) acc[im][in] = (v8f){};

  for (int kb = 0; kb < DD; kb += 16) {
#pragma unroll
    for (int q = 0; q < 8; ++q) {
      const int e = tid + 256 * q;
      const int row = e >> 4, c = e & 15;
      const float x1 = R[(size_t)(i0 + row) * DD + kb + c];
      const float x2 = R[(size_t)(kb + c) * DD + i0 + row];
      Xs[row * 20 + c] = s * (x1 - x2);
      const float w1 = R[(size_t)(kb + c) * DD + j0 + row];
      const float w2 = R[(size_t)(j0 + row) * DD + kb + c];
      Ws[row * 20 + c] = s * (w1 - w2);
    }
    __syncthreads();
    mma_128x128_k16(Xs, Ws, acc, wm, wn, lane);
    __syncthreads();
  }

  const int nlane = lane & 15;
  const int msel = (lane >> 4) << 3;
#pragma unroll
  for (int im = 0; im < 2; ++im)
#pragma unroll
    for (int in = 0; in < 4; ++in)
#pragma unroll
      for (int v = 0; v < 8; ++v)
        Bout[(size_t)(i0 + wm + im * 16 + msel + v) * DD +
             (j0 + wn + in * 16 + nlane)] = acc[im][in][v];
}

// ---- K5: per-task diagonal factor ------------------------------------------
__global__ __launch_bounds__(256) void diag_kernel(
    const float* __restrict__ lora_down, float* __restrict__ ws) {
  const int tid = threadIdx.x;
  const int wave = tid >> 5, lane = tid & 31;
  const int i = blockIdx.x * 8 + wave;
  const int t = blockIdx.y;
  const int* ip = (const int*)(ws + WS_INT);
  float* fac = ws + WS_FAC;
  if (t >= ip[2]) {
    if (lane == 0) fac[(size_t)t * IN_F + i] = 1.0f;
    return;
  }
  const int j = ip[t];
  const int b = i >> 10, ii = i & 1023;
  const float s = ws[WS_SCL + t * 4 + b];
  const float* R = lora_down + (size_t)j * RB * DD * DD + (size_t)b * DD * DD;
  const float* Brow = ws + WS_B + (size_t)(t * 4 + b) * DD * DD + (size_t)ii * DD;
  float sA2 = 0.f, sB2 = 0.f;
  for (int kk = lane; kk < DD; kk += 32) {
    const float a = s * (R[(size_t)ii * DD + kk] - R[(size_t)kk * DD + ii]);
    sA2 += a * a;
    const float bv = Brow[kk];
    sB2 += bv * bv;
  }
  for (int off = 16; off > 0; off >>= 1) {
    sA2 += __shfl_xor(sA2, off);
    sB2 += __shfl_xor(sB2, off);
  }
  if (lane == 0) {
    const float Q = 1.0f + 2.0f * (-sA2 + sB2);
    fac[(size_t)t * IN_F + i] = ws[WS_OMG + j] * Q;
  }
}

// ---- K6: combine into the final diagonal -----------------------------------
__global__ __launch_bounds__(256) void combine_kernel(float* __restrict__ ws) {
  const int i = blockIdx.x * 256 + threadIdx.x;
  ws[WS_DVEC + i] = ws[WS_FAC + i] * ws[WS_FAC + IN_F + i];
}

// ---- K7: out = (X*diag) @ W^T, bf16 hi/lo split, v_wmma_f32_16x16x32_bf16 --
// LDS tiles are stored fragment-linear: each lane's 16-element (32 B) fragment
// is contiguous, matching the 16-bit A(16x32)/B(32x16) WMMA lane layouts.
__global__ __launch_bounds__(256) void main_gemm_kernel(
    const float* __restrict__ X, const float* __restrict__ W,
    const float* __restrict__ dvec, float* __restrict__ out) {
  __shared__ __align__(32) __bf16 XsHi[8 * 32 * 16];   // 8 KB
  __shared__ __align__(32) __bf16 XsLo[8 * 32 * 16];
  __shared__ __align__(32) __bf16 WsHi[8 * 32 * 16];
  __shared__ __align__(32) __bf16 WsLo[8 * 32 * 16];
  const int tid = threadIdx.x;
  const int wave = tid >> 5, lane = tid & 31;
  const int mt0 = (wave & 3) * 2;        // 2 M-tiles of 16
  const int nt0 = (wave >> 2) * 4;       // 4 N-tiles of 16
  const int mBase = blockIdx.y * 128;
  const int nBase = blockIdx.x * 128;

  v8f acc[2][4];
#pragma unroll
  for (int im = 0; im < 2; ++im)
#pragma unroll
    for (int in = 0; in < 4; ++in) acc[im][in] = (v8f){};

  for (int kb = 0; kb < IN_F; kb += 32) {
    // ---- stage hi/lo bf16 tiles, swizzled to fragment-linear order ----
#pragma unroll
    for (int q = 0; q < 4; ++q) {
      const int idx4 = tid + 256 * q;          // 0..1023 float4 slots
      const int row = idx4 >> 3;               // 0..127
      const int c4 = (idx4 & 7) << 2;          // 0,4,...,28
      const float4 xv = *(const float4*)&X[(size_t)(mBase + row) * IN_F + kb + c4];
      const float4 dv = *(const float4*)&dvec[kb + c4];
      const float4 wv = *(const float4*)&W[(size_t)(nBase + row) * IN_F + kb + c4];
      float xs[4] = {xv.x * dv.x, xv.y * dv.y, xv.z * dv.z, xv.w * dv.w};
      float wsv[4] = {wv.x, wv.y, wv.z, wv.w};
      const int mtile = row >> 4, mr = row & 15;
      const int ntile = row >> 4, nr = row & 15;
#pragma unroll
      for (int c = 0; c < 4; ++c) {
        const int k = c4 + c;
        // A swizzle: lane = mr + ((k>>3)&1)*16 ; pos = ((k>>4)<<3) + (k&7)
        const int alane = mr + (((k >> 3) & 1) << 4);
        const int apos = ((k >> 4) << 3) + (k & 7);
        const int aidx = mtile * 512 + alane * 16 + apos;
        const __bf16 xh = (__bf16)xs[c];
        XsHi[aidx] = xh;
        XsLo[aidx] = (__bf16)(xs[c] - (float)xh);
        // B swizzle: lane = nr + ((k>>3)&1)*16 ; pos = ((k>>4)<<3) + (k&7)
        const int blane = nr + (((k >> 3) & 1) << 4);
        const int bpos = ((k >> 4) << 3) + (k & 7);
        const int bidx = ntile * 512 + blane * 16 + bpos;
        const __bf16 wh = (__bf16)wsv[c];
        WsHi[bidx] = wh;
        WsLo[bidx] = (__bf16)(wsv[c] - (float)wh);
      }
    }
    __syncthreads();

    // ---- hi*hi + lo*hi + hi*lo (drop lo*lo) ----
    v16bf aHi[2], bHi[4];
#pragma unroll
    for (int im = 0; im < 2; ++im)
      aHi[im] = *(const v16bf*)&XsHi[((mt0 + im) * 32 + lane) * 16];
#pragma unroll
    for (int in = 0; in < 4; ++in)
      bHi[in] = *(const v16bf*)&WsHi[((nt0 + in) * 32 + lane) * 16];
#pragma unroll
    for (int im = 0; im < 2; ++im)
#pragma unroll
      for (int in = 0; in < 4; ++in)
        acc[im][in] = wmma_bf16(aHi[im], bHi[in], acc[im][in]);

    {
      v16bf aLo[2];
#pragma unroll
      for (int im = 0; im < 2; ++im)
        aLo[im] = *(const v16bf*)&XsLo[((mt0 + im) * 32 + lane) * 16];
#pragma unroll
      for (int im = 0; im < 2; ++im)
#pragma unroll
        for (int in = 0; in < 4; ++in)
          acc[im][in] = wmma_bf16(aLo[im], bHi[in], acc[im][in]);
    }
    {
      v16bf bLo[4];
#pragma unroll
      for (int in = 0; in < 4; ++in)
        bLo[in] = *(const v16bf*)&WsLo[((nt0 + in) * 32 + lane) * 16];
#pragma unroll
      for (int im = 0; im < 2; ++im)
#pragma unroll
        for (int in = 0; in < 4; ++in)
          acc[im][in] = wmma_bf16(aHi[im], bLo[in], acc[im][in]);
    }
    __syncthreads();
  }

  const int nlane = lane & 15;
  const int msel = (lane >> 4) << 3;
#pragma unroll
  for (int im = 0; im < 2; ++im)
#pragma unroll
    for (int in = 0; in < 4; ++in)
#pragma unroll
      for (int v = 0; v < 8; ++v)
        out[(size_t)(mBase + mt0 * 16 + im * 16 + msel + v) * OUT_F +
            (nBase + nt0 * 16 + in * 16 + nlane)] = acc[im][in][v];
}

extern "C" void kernel_launch(void* const* d_in, const int* in_sizes, int n_in,
                              void* d_out, int out_size, void* d_ws,
                              size_t ws_size, hipStream_t stream) {
  (void)in_sizes; (void)n_in; (void)out_size; (void)ws_size;
  const float* X = (const float*)d_in[0];
  const float* lora_route = (const float*)d_in[1];
  const float* lora_down = (const float*)d_in[2];
  const float* W = (const float*)d_in[3];
  const int* task_id = (const int*)d_in[4];
  float* out = (float*)d_out;
  float* ws = (float*)d_ws;

  mean_kernel<<<IN_F / 256, 256, 0, stream>>>(X, ws);
  route_topk_kernel<<<1, 256, 0, stream>>>(lora_route, task_id, ws);
  norm_scale_kernel<<<8, 256, 0, stream>>>(lora_down, ws);
  gemm_B_kernel<<<dim3(8, 8, 8), 256, 0, stream>>>(lora_down, ws);
  diag_kernel<<<dim3(512, 2), 256, 0, stream>>>(lora_down, ws);
  combine_kernel<<<IN_F / 256, 256, 0, stream>>>(ws);
  main_gemm_kernel<<<dim3(32, 32), 256, 0, stream>>>(X, W, ws + WS_DVEC, out);
}
